// CriticModel_54254026883590
// MI455X (gfx1250) — matrix operand order repaired and checked by reading, not compiled, not run on hardware
//
#include <hip/hip_runtime.h>

// ---------------------------------------------------------------------------
// MI455X / gfx1250, wave32. Heavy path: v_wmma_f32_16x16x32_f16.
// Layer sequences are manually unrolled so all activation-tile accesses keep
// LDS address-space inference (ds_load_b128 / ds_store_b16, not flat_*).
//
// d_in flattening assumption (jax pytree: dict keys sorted alphabetically,
// tuples/lists in order):
//   0:dw 1:x 2:x_tau 3:tau 4:ExitIndex
//   critic (5..24): 5:Wout 6:W0 7:W1 8..11:bn_in(g,b,m,v)
//     12..15:bn_out(g,b,m,v) 16..19:bn1 20..23:bn2 24:bout
//   actor (25..44): 25:Wout 26:W0 27:W1 28..31:bn_in 32..35:bn_out
//     36..39:bn1 40..43:bn2 44:bout
// ---------------------------------------------------------------------------

typedef _Float16 h16;
typedef __attribute__((ext_vector_type(8)))  _Float16 v8h;
typedef __attribute__((ext_vector_type(16))) _Float16 v16h;
typedef __attribute__((ext_vector_type(8)))  float    v8f;

#define NTT   64
#define DIMF  256
#define XSTR  (DIMF*(NTT+1))       // 16640 floats per batch elem of x
#define LDA   264                  // padded LDS stride (halves), conflict-free
#define LDW   264
#define BNEPS 1e-6f
#define DTF   (1.0f/64.0f)

#define SMEM_BYTES (size_t)(2*64*LDA*sizeof(h16) + 256*LDW*sizeof(h16) + 64*sizeof(float))

// ---------------- WMMA fragment loaders (wave32, 16x16x32 f16) -------------
// A 16x32 (MxK): lane<16 holds row m=lane, K = k0+{0..7} and k0+16+{0..7};
//                lane>=16 same rows, K offset +8.  -> two ds_load_b128.
__device__ __forceinline__ v16h load_frag_A(const h16* base, int lane, int k0) {
  const int half = lane >> 4;
  const int ml   = lane & 15;
  const h16* p = base + ml * LDA + k0 + half * 8;
  v8h lo = *(const v8h*)p;
  v8h hi = *(const v8h*)(p + 16);
  return __builtin_shufflevector(lo, hi, 0,1,2,3,4,5,6,7,8,9,10,11,12,13,14,15);
}
// B 32x16 (KxN) from W^T[n][k]: lane holds col n = n0 + lane%16,
// K = k0 + 8*half + {0..7} and k0 + 16 + 8*half + {0..7}. -> two ds_load_b128.
__device__ __forceinline__ v16h load_frag_B(const h16* wt, int lane, int k0) {
  const int half = lane >> 4;
  const int nl   = lane & 15;
  const h16* p = wt + nl * LDW + k0 + half * 8;
  v8h lo = *(const v8h*)p;
  v8h hi = *(const v8h*)(p + 16);
  return __builtin_shufflevector(lo, hi, 0,1,2,3,4,5,6,7,8,9,10,11,12,13,14,15);
}

// ------------- 64x256 @ 256x256 tile GEMM (8 waves cooperate) --------------
// wave wid: mi = wid&1 -> rows [mi*32, mi*32+32); ni = wid>>1 -> cols [ni*64,+64)
// acc[2][4]: 2 m-subtiles x 4 n-subtiles of 16x16 f32.
__device__ __forceinline__ void gemm64x256(const h16* actIn, const h16* wlds,
                                           v8f acc[2][4], int lane, int wid) {
  const int mi = wid & 1, ni = wid >> 1;
  const h16* a0b = actIn + (mi * 32) * LDA;
  const h16* a1b = actIn + (mi * 32 + 16) * LDA;
  const h16* wb  = wlds + (ni * 64) * LDW;
#pragma unroll
  for (int kk = 0; kk < 8; ++kk) {
    const int k0 = kk * 32;
    v16h a0 = load_frag_A(a0b, lane, k0);
    v16h a1 = load_frag_A(a1b, lane, k0);
#pragma unroll
    for (int j = 0; j < 4; ++j) {
      v16h bf = load_frag_B(wb + j * 16 * LDW, lane, k0);
      acc[0][j] = __builtin_amdgcn_wmma_f32_16x16x32_f16(false, a0, false, bf,
                                                         (short)0, acc[0][j], false, false);
      acc[1][j] = __builtin_amdgcn_wmma_f32_16x16x32_f16(false, a1, false, bf,
                                                         (short)0, acc[1][j], false, false);
    }
  }
}

__device__ __forceinline__ void stage_weights(h16* wlds, const h16* __restrict__ gw, int tid) {
#pragma unroll 4
  for (int c = tid; c < 8192; c += 256) {   // 8192 chunks of 8 halves (16B), coalesced
    const int n = c >> 5, k = (c & 31) << 3;
    *(v8h*)(wlds + n * LDW + k) = *(const v8h*)(gw + n * DIMF + k);
  }
}

// C/D layout: lane holds col n = n0 + lane%16; VGPR r holds row m0 + r + 8*(lane>=16)
__device__ __forceinline__ void epilogue_relu(v8f acc[2][4], h16* actOut,
                                              const float* __restrict__ sc,
                                              const float* __restrict__ sh,
                                              int lane, int wid) {
  const int mi = wid & 1, ni = wid >> 1;
  const int half = lane >> 4, nl = lane & 15;
#pragma unroll
  for (int t2 = 0; t2 < 2; ++t2) {
    const int mrow = mi * 32 + t2 * 16 + half * 8;
#pragma unroll
    for (int j = 0; j < 4; ++j) {
      const int n = ni * 64 + j * 16 + nl;
      const float s = sc[n], h = sh[n];
#pragma unroll
      for (int r = 0; r < 8; ++r) {
        float v = acc[t2][j][r] * s + h;
        v = v > 0.f ? v : 0.f;
        actOut[(mrow + r) * LDA + n] = (h16)v;
      }
    }
  }
}

__device__ __forceinline__ void epilogue_u2(v8f acc[2][4],
                                            const float* __restrict__ sc,
                                            const float* __restrict__ sh,
                                            float* wsum, int lane, int wid) {
  const int mi = wid & 1, ni = wid >> 1;
  const int half = lane >> 4, nl = lane & 15;
#pragma unroll
  for (int t2 = 0; t2 < 2; ++t2) {
    float ps[8] = {0.f,0.f,0.f,0.f,0.f,0.f,0.f,0.f};
#pragma unroll
    for (int j = 0; j < 4; ++j) {
      const int n = ni * 64 + j * 16 + nl;
      const float s = sc[n], h = sh[n];
#pragma unroll
      for (int r = 0; r < 8; ++r) {
        const float v = acc[t2][j][r] * s + h;   // u element, no relu
        ps[r] += v * v;
      }
    }
#pragma unroll
    for (int r = 0; r < 8; ++r) {
      float p = ps[r];                           // reduce over 16 lanes (same half)
      p += __shfl_xor(p, 1, 32);
      p += __shfl_xor(p, 2, 32);
      p += __shfl_xor(p, 4, 32);
      p += __shfl_xor(p, 8, 32);
      if (nl == 0) atomicAdd(&wsum[mi * 32 + t2 * 16 + half * 8 + r], p);
    }
  }
}

// ------------------------------- prep kernel -------------------------------
struct PrepArgs {
  const float* srcW[5];   // a_W0, a_W1, a_Wout, c_W0, c_W1 (256x256, row-major k,n)
  h16*         dstW[5];   // transposed f16 W^T[n][k], stride 256
  const float* g[7]; const float* bb[7]; const float* m[7]; const float* v[7];
  const float* a_bout;    // folded into bn#3 shift
  float* scale; float* shift;   // 7*256 each
};

__global__ void prep_kernel(PrepArgs p) {
  const int blk = blockIdx.x, t = threadIdx.x;
  if (blk < 5 * 256) {
    const int mid = blk >> 8, kr = blk & 255;
    const float* src = p.srcW[mid] + (size_t)kr * 256;
    p.dstW[mid][(size_t)t * 256 + kr] = (h16)src[t];
  } else {
    const int bn = blk - 5 * 256;               // 0:a_in 1:a_1 2:a_2 3:a_out 4:c_in 5:c_1 6:c_2
    const float s  = p.g[bn][t] * rsqrtf(p.v[bn][t] + BNEPS);
    float sh = p.bb[bn][t] - p.m[bn][t] * s;
    if (bn == 3) sh += s * p.a_bout[t];         // fold actor output bias
    p.scale[bn * 256 + t] = s;
    p.shift[bn * 256 + t] = sh;
  }
}

// ------------------------------ actor kernel -------------------------------
struct ActorArgs {
  const float* x;                 // (4096,256,65)
  const h16* wt0; const h16* wt1; const h16* wt2;  // W0^T, W1^T, Wout^T (f16)
  const float* scale; const float* shift;          // rows 0..3 used
  float* w_out;                   // (4096,64)
};

__global__ void __launch_bounds__(256, 1) actor_kernel(ActorArgs p) {
  extern __shared__ char smem[];
  h16*   act0 = (h16*)smem;
  h16*   act1 = act0 + 64 * LDA;
  h16*   wlds = act1 + 64 * LDA;
  float* wsum = (float*)(wlds + 256 * LDW);

  const int tid = threadIdx.x, lane = tid & 31, wid = tid >> 5;
  const int b = blockIdx.x;

  if (tid < 64) wsum[tid] = 0.f;

  { // load x[b,:,0:64]: coalesced over t, transpose into act0 with bn_in; ||xt||^2
    const float* sc = p.scale, *sh = p.shift;
    const float* xb = p.x + (size_t)b * XSTR;
    const int t = tid & 63, dg = tid >> 6;
    float ss = 0.f;
    for (int d = dg; d < DIMF; d += 4) {
      const float v = xb[(size_t)d * (NTT + 1) + t];
      ss += v * v;
      act0[t * LDA + d] = (h16)(sc[d] * v + sh[d]);
    }
    stage_weights(wlds, p.wt0, tid);     // overlap W0 staging with input stage
    __syncthreads();                     // wsum zeroed before atomics (tid<64 path above)
    atomicAdd(&wsum[t], ss);
  }
  __syncthreads();

  // ---- layer 0: act0 @ W0 -> bn1+relu -> act1 (direct LDS pointers) ----
  {
    v8f acc[2][4] = {};
    gemm64x256(act0, wlds, acc, lane, wid);
    epilogue_relu(acc, act1, p.scale + 1 * 256, p.shift + 1 * 256, lane, wid);
  }
  __syncthreads();
  stage_weights(wlds, p.wt1, tid);
  __syncthreads();

  // ---- layer 1: act1 @ W1 -> bn2+relu -> act0 ----
  {
    v8f acc[2][4] = {};
    gemm64x256(act1, wlds, acc, lane, wid);
    epilogue_relu(acc, act0, p.scale + 2 * 256, p.shift + 2 * 256, lane, wid);
  }
  __syncthreads();
  stage_weights(wlds, p.wt2, tid);
  __syncthreads();

  // ---- layer 2: act0 @ Wout -> bn_out (+bout folded) -> sum u^2 ----
  {
    v8f acc[2][4] = {};
    gemm64x256(act0, wlds, acc, lane, wid);
    epilogue_u2(acc, p.scale + 3 * 256, p.shift + 3 * 256, wsum, lane, wid);
  }

  __syncthreads();
  if (tid < 64) p.w_out[(size_t)b * NTT + tid] = wsum[tid];
}

// ------------------------------ critic kernel ------------------------------
struct CriticArgs {
  const float* x;                 // for x[:,:,0] rows
  const float* x_tau;             // (4096,256)
  const h16* wt0; const h16* wt1; // cW0^T, cW1^T
  const float* wout;              // c_Wout (256 f32)
  const float* scale; const float* shift;   // rows 4..6 used
  const float* g_o; const float* b_o; const float* m_o; const float* v_o; const float* bout;
  float* v_out;                   // [0..4095]=v_tau, [4096..8191]=v0
  float* Z;                       // (4096)
};

__global__ void __launch_bounds__(256, 1) critic_kernel(CriticArgs p) {
  extern __shared__ char smem[];
  h16*   act0 = (h16*)smem;
  h16*   act1 = act0 + 64 * LDA;
  h16*   wlds = act1 + 64 * LDA;
  float* wsum = (float*)(wlds + 256 * LDW);

  const int tid = threadIdx.x, lane = tid & 31, wid = tid >> 5;
  const bool is_tau = blockIdx.x < 64;
  const int  r0 = (is_tau ? blockIdx.x : blockIdx.x - 64) * 64;

  if (tid < 64) wsum[tid] = 0.f;
  __syncthreads();

  { // load 64 rows, bn_in, f16 to act0; Z accumulation for x_tau blocks
    const float* sc = p.scale + 4 * 256;
    const float* sh = p.shift + 4 * 256;
    const int d = tid;                     // one feature per thread, 64 rows
    const float scd = sc[d], shd = sh[d];
    for (int r = 0; r < 64; ++r) {
      const float v = is_tau ? p.x_tau[(size_t)(r0 + r) * DIMF + d]
                             : p.x[(size_t)(r0 + r) * XSTR + (size_t)d * (NTT + 1)];
      act0[r * LDA + d] = (h16)(scd * v + shd);
      if (is_tau) {
        float v2 = v * v;
        v2 += __shfl_xor(v2, 1, 32);  v2 += __shfl_xor(v2, 2, 32);
        v2 += __shfl_xor(v2, 4, 32);  v2 += __shfl_xor(v2, 8, 32);
        v2 += __shfl_xor(v2, 16, 32);
        if (lane == 0) atomicAdd(&wsum[r], v2);
      }
    }
    stage_weights(wlds, p.wt0, tid);
  }
  __syncthreads();

  // ---- layer 0: act0 @ cW0 -> bn1+relu -> act1 ----
  {
    v8f acc[2][4] = {};
    gemm64x256(act0, wlds, acc, lane, wid);
    epilogue_relu(acc, act1, p.scale + 5 * 256, p.shift + 5 * 256, lane, wid);
  }
  __syncthreads();
  stage_weights(wlds, p.wt1, tid);
  __syncthreads();

  // ---- layer 1: act1 @ cW1 -> bn2+relu -> act0 ----
  {
    v8f acc[2][4] = {};
    gemm64x256(act1, wlds, acc, lane, wid);
    epilogue_relu(acc, act0, p.scale + 6 * 256, p.shift + 6 * 256, lane, wid);
  }
  __syncthreads();

  { // final 256 -> 1 dense + bn_out (activations in act0)
    const int r = tid >> 2, q = tid & 3;
    const h16* arow = act0 + r * LDA;
    float part = 0.f;
    for (int d = q * 64; d < q * 64 + 64; ++d) part += (float)arow[d] * p.wout[d];
    part += __shfl_xor(part, 1, 32);
    part += __shfl_xor(part, 2, 32);
    if (q == 0) {
      const float g = *p.g_o, bb = *p.b_o, mm = *p.m_o, vv = *p.v_o, bo = *p.bout;
      const float s = g * rsqrtf(vv + BNEPS);
      p.v_out[(is_tau ? 0 : 4096) + r0 + r] = s * (part + bo - mm) + bb;
    }
    if (is_tau && tid < 64) p.Z[r0 + tid] = wsum[tid];
  }
}

// ------------------------------- final kernel ------------------------------
__global__ void final_kernel(const float* __restrict__ w, const float* __restrict__ tau,
                             const int* __restrict__ ei, const float* __restrict__ Z,
                             const float* __restrict__ v, float* __restrict__ out) {
  __shared__ float red[2];
  const int b = blockIdx.x, t = threadIdx.x;   // 64 threads
  const int e = ei[b];
  const float wt = (t < e) ? DTF : ((t == e) ? (tau[b] - (float)t * DTF) : 0.f);
  float par = w[(size_t)b * NTT + t] * wt;
  par += __shfl_xor(par, 1, 32);  par += __shfl_xor(par, 2, 32);
  par += __shfl_xor(par, 4, 32);  par += __shfl_xor(par, 8, 32);
  par += __shfl_xor(par, 16, 32);
  if ((t & 31) == 0) red[t >> 5] = par;
  __syncthreads();
  if (t == 0) {
    float y = red[0] + red[1];
    y += (e < NTT) ? Z[b] : v[b];     // v[b] = v_tau
    out[b] = y - v[4096 + b];         // v0
  }
}

// ------------------------------ kernel_launch ------------------------------
extern "C" void kernel_launch(void* const* d_in, const int* in_sizes, int n_in,
                              void* d_out, int out_size, void* d_ws, size_t ws_size,
                              hipStream_t stream) {
  const float* x     = (const float*)d_in[1];
  const float* x_tau = (const float*)d_in[2];
  const float* tau   = (const float*)d_in[3];
  const int*   ei    = (const int*)d_in[4];

  const float* c_Wout = (const float*)d_in[5];
  const float* c_W0   = (const float*)d_in[6];
  const float* c_W1   = (const float*)d_in[7];
  const float* a_Wout = (const float*)d_in[25];
  const float* a_W0   = (const float*)d_in[26];
  const float* a_W1   = (const float*)d_in[27];

  // workspace carve
  char* ws = (char*)d_ws;
  h16* wtA0 = (h16*)(ws);                       // 5 x 128 KB transposed f16 weights
  h16* wtA1 = (h16*)(ws + 131072);
  h16* wtA2 = (h16*)(ws + 262144);
  h16* wtC0 = (h16*)(ws + 393216);
  h16* wtC1 = (h16*)(ws + 524288);
  float* scale = (float*)(ws + 655360);         // 7*256
  float* shift = (float*)(ws + 662528);         // 7*256
  float* wbuf  = (float*)(ws + 669696);         // 4096*64
  float* vbuf  = (float*)(ws + 1718272);        // 8192 (v_tau | v0)
  float* Zbuf  = (float*)(ws + 1751040);        // 4096

  PrepArgs pp;
  pp.srcW[0] = a_W0;  pp.dstW[0] = wtA0;
  pp.srcW[1] = a_W1;  pp.dstW[1] = wtA1;
  pp.srcW[2] = a_Wout;pp.dstW[2] = wtA2;
  pp.srcW[3] = c_W0;  pp.dstW[3] = wtC0;
  pp.srcW[4] = c_W1;  pp.dstW[4] = wtC1;
  // bn order: 0:a_in 1:a_1 2:a_2 3:a_out 4:c_in 5:c_1 6:c_2
  const int bnbase[7] = {28, 36, 40, 32, 8, 16, 20};
  for (int i = 0; i < 7; ++i) {
    pp.g[i]  = (const float*)d_in[bnbase[i] + 0];
    pp.bb[i] = (const float*)d_in[bnbase[i] + 1];
    pp.m[i]  = (const float*)d_in[bnbase[i] + 2];
    pp.v[i]  = (const float*)d_in[bnbase[i] + 3];
  }
  pp.a_bout = (const float*)d_in[44];
  pp.scale = scale; pp.shift = shift;

  ActorArgs aa;
  aa.x = x; aa.wt0 = wtA0; aa.wt1 = wtA1; aa.wt2 = wtA2;
  aa.scale = scale; aa.shift = shift; aa.w_out = wbuf;

  CriticArgs ca;
  ca.x = x; ca.x_tau = x_tau; ca.wt0 = wtC0; ca.wt1 = wtC1;
  ca.wout = c_Wout; ca.scale = scale; ca.shift = shift;
  ca.g_o = (const float*)d_in[12]; ca.b_o = (const float*)d_in[13];
  ca.m_o = (const float*)d_in[14]; ca.v_o = (const float*)d_in[15];
  ca.bout = (const float*)d_in[24];
  ca.v_out = vbuf; ca.Z = Zbuf;

  (void)hipFuncSetAttribute((const void*)actor_kernel,
                            hipFuncAttributeMaxDynamicSharedMemorySize, (int)SMEM_BYTES);
  (void)hipFuncSetAttribute((const void*)critic_kernel,
                            hipFuncAttributeMaxDynamicSharedMemorySize, (int)SMEM_BYTES);

  prep_kernel  <<<5 * 256 + 7, 256, 0, stream>>>(pp);
  critic_kernel<<<128, 256, SMEM_BYTES, stream>>>(ca);
  actor_kernel <<<4096, 256, SMEM_BYTES, stream>>>(aa);
  final_kernel <<<4096, 64, 0, stream>>>(wbuf, tau, ei, Zbuf, vbuf, (float*)d_out);
}
